// MultiHeadAttention_84293028151418
// MI455X (gfx1250) — matrix operand order repaired and checked by reading, not compiled, not run on hardware
//
#include <hip/hip_runtime.h>
#include <hip/hip_bf16.h>

#define EMBED 512
#define NHEAD 8
#define DEPTH 64
#define BATCH 2
#define SEQ   4096

// log2(e) folded into the softmax so v_exp_f32 (native exp2) is used directly.
#define LOG2E 1.4426950408889634f

typedef __bf16 bf16_t;
typedef __attribute__((ext_vector_type(16))) __bf16 v16bf;
typedef __attribute__((ext_vector_type(8)))  __bf16 v8bf;
typedef __attribute__((ext_vector_type(8)))  float  v8f;
typedef __attribute__((ext_vector_type(4)))  float  v4f;

static __device__ inline v8f wmma_bf16(v16bf a, v16bf b, v8f c) {
  return __builtin_amdgcn_wmma_f32_16x16x32_bf16(false, a, false, b, (short)0, c, false, false);
}

static __device__ inline v8f vzero8() {
  v8f z;
#pragma unroll
  for (int i = 0; i < 8; ++i) z[i] = 0.0f;
  return z;
}

// ---- CDNA5 async global->LDS copy (ISA §10.7 / §15.18.3) -------------------
// Per-lane 16-byte transfer written directly into LDS, tracked by ASYNCcnt.
static __device__ inline void async_copy_b128(void* lds_dst, const void* gsrc) {
  unsigned lds_addr = (unsigned)(uintptr_t)lds_dst;   // addr[31:0] = LDS offset
  asm volatile("global_load_async_to_lds_b128 %0, %1, off"
               :
               : "v"(lds_addr), "v"(gsrc)
               : "memory");
}

static __device__ inline void wait_async0() {
#if __has_builtin(__builtin_amdgcn_s_wait_asynccnt)
  __builtin_amdgcn_s_wait_asynccnt(0);
#else
  asm volatile("s_wait_asynccnt 0x0" ::: "memory");
#endif
}

// ---- fragment loaders ------------------------------------------------------
// A 16x32 bf16 (ISA 7.12.2): lane L holds row M=L%16; elems 0..7 -> K = kg*8+e,
// elems 8..15 -> K = 16+kg*8+(e-8), kg = L/16.
static __device__ inline v16bf load_a_f32(const float* base, int lda, int mBase,
                                          int kBase, int lane) {
  int row = mBase + (lane & 15);
  int kg  = lane >> 4;
  const float* p = base + (size_t)row * lda + kBase + kg * 8;
  v4f x0 = *(const v4f*)(p);
  v4f x1 = *(const v4f*)(p + 4);
  v4f x2 = *(const v4f*)(p + 16);
  v4f x3 = *(const v4f*)(p + 20);
  v16bf a;
#pragma unroll
  for (int e = 0; e < 4; ++e) {
    a[e]      = (bf16_t)x0[e];
    a[4 + e]  = (bf16_t)x1[e];
    a[8 + e]  = (bf16_t)x2[e];
    a[12 + e] = (bf16_t)x3[e];
  }
  return a;
}

static __device__ inline v16bf load_a_bf16(const bf16_t* base, int lda, int mBase,
                                           int kBase, int lane) {
  int row = mBase + (lane & 15);
  int kg  = lane >> 4;
  const bf16_t* p = base + (size_t)row * lda + kBase + kg * 8;
  v8bf lo = *(const v8bf*)(p);
  v8bf hi = *(const v8bf*)(p + 16);
  v16bf a;
#pragma unroll
  for (int e = 0; e < 8; ++e) { a[e] = lo[e]; a[e + 8] = hi[e]; }
  return a;
}

// B 32x16 bf16: lane L holds col N=L%16; elems 0..7 -> K = (L/16)*8+e, elems 8..15 -> K = 16+(L/16)*8+(e-8).
// base laid out [col][k] (K-contiguous per column); works for global or LDS.
static __device__ inline v16bf load_b_kc(const bf16_t* base, int ldk, int nBase,
                                         int kBase, int lane) {
  int col = nBase + (lane & 15);
  int k0  = kBase + (lane >> 4) * 8;
  const bf16_t* p = base + (size_t)col * ldk + k0;
  v8bf lo = *(const v8bf*)(p);
  v8bf hi = *(const v8bf*)(p + 16);
  v16bf b;
#pragma unroll
  for (int e = 0; e < 8; ++e) { b[e] = lo[e]; b[e + 8] = hi[e]; }
  return b;
}

// ---- 16-lane butterfly reductions (stay inside lane groups 0-15 / 16-31) ---
static __device__ inline float redmax16(float v) {
  v = fmaxf(v, __shfl_xor(v, 1));
  v = fmaxf(v, __shfl_xor(v, 2));
  v = fmaxf(v, __shfl_xor(v, 4));
  v = fmaxf(v, __shfl_xor(v, 8));
  return v;
}
static __device__ inline float redsum16(float v) {
  v += __shfl_xor(v, 1);
  v += __shfl_xor(v, 2);
  v += __shfl_xor(v, 4);
  v += __shfl_xor(v, 8);
  return v;
}

// ---- kernel 1: pack fp32 weights into transposed bf16 ----------------------
__global__ void mha_packw(const float* __restrict__ W0, const float* __restrict__ W1,
                          const float* __restrict__ W2, const float* __restrict__ W3,
                          bf16_t* __restrict__ T0, bf16_t* __restrict__ T1,
                          bf16_t* __restrict__ T2, bf16_t* __restrict__ T3) {
  int i = blockIdx.x * blockDim.x + threadIdx.x;   // 0 .. E*E-1
  int n = i >> 9;
  int k = i & 511;
  const float* W;
  bf16_t* T;
  switch (blockIdx.y) {
    case 0: W = W0; T = T0; break;
    case 1: W = W1; T = T1; break;
    case 2: W = W2; T = T2; break;
    default: W = W3; T = T3; break;
  }
  T[i] = (bf16_t)W[(size_t)k * EMBED + n];
}

// ---- kernel 2: QKV projection GEMM -----------------------------------------
// out = X[B*S,E] @ W + b ; mode 0: Q head-layout pre-scaled by log2e/sqrt(D),
// mode 1: K head-layout [B,H,S,D], mode 2: V transposed [B,H,D,S].
__global__ void mha_proj(const float* __restrict__ X, const bf16_t* __restrict__ WT,
                         const float* __restrict__ bias, bf16_t* __restrict__ out,
                         int mode) {
  int lane  = threadIdx.x & 31;
  int wave  = threadIdx.x >> 5;
  int mTile = blockIdx.x;
  int nTile = blockIdx.y * 4 + wave;

  v8f acc = vzero8();
#pragma unroll 4
  for (int kb = 0; kb < EMBED; kb += 32) {
    v16bf a = load_a_f32(X, EMBED, mTile * 16, kb, lane);
    v16bf b = load_b_kc(WT, EMBED, nTile * 16, kb, lane);
    acc = wmma_bf16(a, b, acc);
  }

  int n  = nTile * 16 + (lane & 15);
  float bv = bias[n];
  int h = n >> 6, d = n & 63;
#pragma unroll
  for (int r = 0; r < 8; ++r) {
    int m    = mTile * 16 + ((lane >> 4) << 3) + r;
    int bidx = m >> 12;        // / SEQ
    int s    = m & (SEQ - 1);
    float val = acc[r] + bv;
    if (mode == 0) val *= 0.125f * LOG2E;   // 1/sqrt(D) * log2(e)
    size_t idx;
    if (mode == 2)
      idx = (((size_t)(bidx * NHEAD + h)) * DEPTH + d) * SEQ + s;
    else
      idx = (((size_t)(bidx * NHEAD + h)) * SEQ + s) * DEPTH + d;
    out[idx] = (bf16_t)val;
  }
}

// ---- kernel 3: flash attention ---------------------------------------------
// 8 waves/block share one (b,h). K/V tiles are double-buffered in LDS and
// filled with GLOBAL_LOAD_ASYNC_TO_LDS_B128 (ASYNCcnt-tracked, no VGPR
// round-trip); next tile's copy runs concurrently with WMMA+softmax on the
// current tile. Softmax in exp2 domain. One wave per 16-row q-tile.
__global__ void __launch_bounds__(256)
mha_flash(const bf16_t* __restrict__ Qh, const bf16_t* __restrict__ Kh,
          const bf16_t* __restrict__ Vt, const float* __restrict__ mask,
          bf16_t* __restrict__ Ctx) {
  __shared__ bf16_t ktile[2][32 * 64];     // [buf][krow][d]   2 x 4 KB
  __shared__ bf16_t vtile[2][64 * 32];     // [buf][d][krow]   2 x 4 KB
  __shared__ bf16_t pshare[8 * 16 * 32];   // per-wave 16x32 P staging, 8 KB

  int tid  = threadIdx.x;
  int lane = tid & 31;
  int wave = tid >> 5;
  // 512 blocks: 16 (b,h) x 32 groups of 8 q-tiles
  int bh    = blockIdx.x >> 5;
  int grp   = blockIdx.x & 31;
  int qTile = grp * 8 + wave;
  int b = bh >> 3, h = bh & 7;

  const bf16_t* Qb = Qh + (size_t)bh * SEQ * DEPTH;
  const bf16_t* Kb = Kh + (size_t)bh * SEQ * DEPTH;
  const bf16_t* Vb = Vt + (size_t)bh * DEPTH * SEQ;
  const float*  mb = mask + (size_t)b * SEQ;
  bf16_t* pb = pshare + wave * 16 * 32;

  // per-thread 16B slots of the cooperative K/V tile copy
  const int krow = tid >> 3, kseg = tid & 7;   // K: 32 rows x 8 segs of 8 bf16
  const int vrow = tid >> 2, vseg = tid & 3;   // V: 64 rows x 4 segs of 8 bf16

  v16bf qa0 = load_a_bf16(Qb, DEPTH, qTile * 16, 0,  lane);
  v16bf qa1 = load_a_bf16(Qb, DEPTH, qTile * 16, 32, lane);

  v8f o0 = vzero8(), o1 = vzero8(), o2 = vzero8(), o3 = vzero8();
  float m_i[8], l_i[8];
#pragma unroll
  for (int r = 0; r < 8; ++r) { m_i[r] = -1e30f; l_i[r] = 0.0f; }

  // prologue: async-copy tile 0 into buffer 0
  async_copy_b128(&ktile[0][krow * 64 + kseg * 8],
                  Kb + (size_t)krow * DEPTH + kseg * 8);
  async_copy_b128(&vtile[0][vrow * 32 + vseg * 8],
                  Vb + (size_t)vrow * SEQ + vseg * 8);
  wait_async0();
  __syncthreads();

  for (int it = 0; it < SEQ / 32; ++it) {
    const int kb  = it * 32;
    const int cur = it & 1;
    const int nxt = cur ^ 1;
    const bf16_t* kt = ktile[cur];
    const bf16_t* vt = vtile[cur];

    if (kb + 32 < SEQ) {   // async-fill the other buffer while we compute
      async_copy_b128(&ktile[nxt][krow * 64 + kseg * 8],
                      Kb + (size_t)(kb + 32 + krow) * DEPTH + kseg * 8);
      async_copy_b128(&vtile[nxt][vrow * 32 + vseg * 8],
                      Vb + (size_t)vrow * SEQ + (kb + 32) + vseg * 8);
      if (kb + 64 < SEQ) {   // prefetch tile i+2 into caches
        __builtin_prefetch(Kb + (size_t)(kb + 64 + krow) * DEPTH + kseg * 8, 0, 0);
        __builtin_prefetch(Vb + (size_t)vrow * SEQ + (kb + 64) + vseg * 8, 0, 0);
      }
    }

    // S = Q @ K^T for 32 key rows (two 16-wide N tiles), K tile from LDS
    v8f s0 = vzero8(), s1 = vzero8();
    s0 = wmma_bf16(qa0, load_b_kc(kt, DEPTH, 0,  0,  lane), s0);
    s0 = wmma_bf16(qa1, load_b_kc(kt, DEPTH, 0,  32, lane), s0);
    s1 = wmma_bf16(qa0, load_b_kc(kt, DEPTH, 16, 0,  lane), s1);
    s1 = wmma_bf16(qa1, load_b_kc(kt, DEPTH, 16, 32, lane), s1);

    float mk0 = mb[kb +      (lane & 15)] * (-1e9f * LOG2E);
    float mk1 = mb[kb + 16 + (lane & 15)] * (-1e9f * LOG2E);

    float alpha[8];
#pragma unroll
    for (int r = 0; r < 8; ++r) {
      float a0 = s0[r] + mk0;
      float a1 = s1[r] + mk1;
      float t  = redmax16(fmaxf(a0, a1));
      float mn = fmaxf(m_i[r], t);
      alpha[r] = exp2f(m_i[r] - mn);     // native v_exp_f32
      m_i[r]   = mn;
      float p0 = exp2f(a0 - mn);
      float p1 = exp2f(a1 - mn);
      s0[r] = p0;
      s1[r] = p1;
      l_i[r] = l_i[r] * alpha[r] + redsum16(p0 + p1);
    }
#pragma unroll
    for (int r = 0; r < 8; ++r) {
      o0[r] *= alpha[r]; o1[r] *= alpha[r];
      o2[r] *= alpha[r]; o3[r] *= alpha[r];
    }

    // C-layout (col per lane) -> A-fragment (row per lane) via per-wave LDS
    {
      int mrow = (lane >> 4) << 3;
      int n    = lane & 15;
#pragma unroll
      for (int r = 0; r < 8; ++r) {
        pb[(mrow + r) * 32 + n]      = (bf16_t)s0[r];
        pb[(mrow + r) * 32 + 16 + n] = (bf16_t)s1[r];
      }
    }
    v16bf pa;
    {
      int row = lane & 15;
      int kg  = lane >> 4;
      const bf16_t* pp = pb + row * 32 + kg * 8;
      v8bf lo = *(const v8bf*)(pp);
      v8bf hi = *(const v8bf*)(pp + 16);
#pragma unroll
      for (int e = 0; e < 8; ++e) { pa[e] = lo[e]; pa[e + 8] = hi[e]; }
    }

    // O += P @ V, V tile from LDS ([d][krow], K-contiguous per column)
    o0 = wmma_bf16(pa, load_b_kc(vt, 32, 0,  0, lane), o0);
    o1 = wmma_bf16(pa, load_b_kc(vt, 32, 16, 0, lane), o1);
    o2 = wmma_bf16(pa, load_b_kc(vt, 32, 32, 0, lane), o2);
    o3 = wmma_bf16(pa, load_b_kc(vt, 32, 48, 0, lane), o3);

    // own async copies landed + all waves done with this buffer
    wait_async0();
    __syncthreads();
  }

  // normalize and write context back in [B,S,E] layout (bf16)
  int ncol = lane & 15;
#pragma unroll
  for (int r = 0; r < 8; ++r) {
    int srow  = qTile * 16 + ((lane >> 4) << 3) + r;
    float inv = 1.0f / l_i[r];
    size_t rowoff = ((size_t)b * SEQ + srow) * EMBED + h * DEPTH;
    Ctx[rowoff + 0  + ncol] = (bf16_t)(o0[r] * inv);
    Ctx[rowoff + 16 + ncol] = (bf16_t)(o1[r] * inv);
    Ctx[rowoff + 32 + ncol] = (bf16_t)(o2[r] * inv);
    Ctx[rowoff + 48 + ncol] = (bf16_t)(o3[r] * inv);
  }
}

// ---- kernel 4: output projection (bf16 x bf16 -> fp32 + bias) --------------
__global__ void mha_outproj(const bf16_t* __restrict__ Ctx, const bf16_t* __restrict__ WoT,
                            const float* __restrict__ bias, float* __restrict__ out) {
  int lane  = threadIdx.x & 31;
  int wave  = threadIdx.x >> 5;
  int mTile = blockIdx.x;
  int nTile = blockIdx.y * 4 + wave;

  v8f acc = vzero8();
#pragma unroll 4
  for (int kb = 0; kb < EMBED; kb += 32) {
    v16bf a = load_a_bf16(Ctx, EMBED, mTile * 16, kb, lane);
    v16bf b = load_b_kc(WoT, EMBED, nTile * 16, kb, lane);
    acc = wmma_bf16(a, b, acc);
  }

  int n   = nTile * 16 + (lane & 15);
  float bv = bias[n];
#pragma unroll
  for (int r = 0; r < 8; ++r) {
    int m = mTile * 16 + ((lane >> 4) << 3) + r;
    out[(size_t)m * EMBED + n] = acc[r] + bv;
  }
}

extern "C" void kernel_launch(void* const* d_in, const int* in_sizes, int n_in,
                              void* d_out, int out_size, void* d_ws, size_t ws_size,
                              hipStream_t stream) {
  const float* q    = (const float*)d_in[0];
  const float* k    = (const float*)d_in[1];
  const float* v    = (const float*)d_in[2];
  const float* mask = (const float*)d_in[3];
  const float* Wq   = (const float*)d_in[4];
  const float* bq   = (const float*)d_in[5];
  const float* Wk   = (const float*)d_in[6];
  const float* bk   = (const float*)d_in[7];
  const float* Wv   = (const float*)d_in[8];
  const float* bv   = (const float*)d_in[9];
  const float* Wo   = (const float*)d_in[10];
  const float* bo   = (const float*)d_in[11];

  char* ws = (char*)d_ws;
  const size_t WSZ   = (size_t)EMBED * EMBED * sizeof(bf16_t);               // 512 KB
  const size_t HDSZ  = (size_t)BATCH * NHEAD * SEQ * DEPTH * sizeof(bf16_t); // 8 MB
  bf16_t* WqT = (bf16_t*)(ws);
  bf16_t* WkT = (bf16_t*)(ws + WSZ);
  bf16_t* WvT = (bf16_t*)(ws + 2 * WSZ);
  bf16_t* WoT = (bf16_t*)(ws + 3 * WSZ);
  bf16_t* Qh  = (bf16_t*)(ws + 4 * WSZ);
  bf16_t* Kh  = (bf16_t*)(ws + 4 * WSZ + HDSZ);
  bf16_t* Vt  = (bf16_t*)(ws + 4 * WSZ + 2 * HDSZ);
  bf16_t* Ctx = (bf16_t*)(ws + 4 * WSZ + 3 * HDSZ);

  // 1) pack all four weight matrices (transposed, bf16)
  mha_packw<<<dim3(EMBED * EMBED / 256, 4), 256, 0, stream>>>(Wq, Wk, Wv, Wo,
                                                              WqT, WkT, WvT, WoT);
  // 2) projections: M tiles = B*S/16 = 512, N tiles = 32 (4 waves/block)
  mha_proj<<<dim3(512, 8), 128, 0, stream>>>(q, WqT, bq, Qh, 0);
  mha_proj<<<dim3(512, 8), 128, 0, stream>>>(k, WkT, bk, Kh, 1);
  mha_proj<<<dim3(512, 8), 128, 0, stream>>>(v, WvT, bv, Vt, 2);
  // 3) flash attention: 512 blocks x 8 waves, async double-buffered K/V in LDS
  mha_flash<<<dim3(512), 256, 0, stream>>>(Qh, Kh, Vt, mask, Ctx);
  // 4) output projection -> fp32 d_out
  mha_outproj<<<dim3(512, 8), 128, 0, stream>>>(Ctx, WoT, bo, (float*)d_out);
}
